// TraceLegS_11708080849233
// MI455X (gfx1250) — matrix operand order repaired and checked by reading, not compiled, not run on hardware
//
#include <hip/hip_runtime.h>
#include <hip/hip_bf16.h>

typedef __attribute__((ext_vector_type(16))) __bf16 v16bf;
typedef __attribute__((ext_vector_type(8)))  float  v8f;
typedef __attribute__((ext_vector_type(4)))  float  v4f;

namespace {
constexpr int kT    = 1024;  // timesteps
constexpr int kSeq  = 128;   // B*P independent recurrences
constexpr int kH    = 128;   // state width
constexpr int kSlab = 16;    // sequences per wave
constexpr int kChunk   = 32;          // timesteps per chunk
constexpr int kNChunks = kT / kChunk; // 32
constexpr size_t kMat  = (size_t)kSeq * kH;  // floats per timestep (16384)
// d_ws float layout: [A^32 | E[0..31] | carry[0..31]]
constexpr size_t kWsA32   = 0;
constexpr size_t kWsE     = kMat;
constexpr size_t kWsCarry = kMat + (size_t)kNChunks * kMat;
constexpr size_t kWsFloats = kMat + 2 * (size_t)kNChunks * kMat;  // ~4.3 MB
}

// ---- Aᵀ-style matrix -> WMMA B-operand register tiles (bf16) -----------
// Dense 32x16 16-bit B layout: lane l -> column (l&15); element e -> row
// K = kb*32 + (l>>4)*8 + e (e<8), kb*32 + 16 + (l>>4)*8 + (e-8) (e>=8).  Value = Mᵀ[K][j] = M[j][K].
__device__ __forceinline__ void load_breg(const float* __restrict__ M,
                                          int lane, v16bf (&breg)[4][8])
{
    const int half = lane >> 4, col = lane & 15;
    #pragma unroll
    for (int kb = 0; kb < 4; ++kb)
        #pragma unroll
        for (int jb = 0; jb < 8; ++jb) {
            const float* src = M + (size_t)(jb * 16 + col) * kH + kb * 32 + half * 8;
            v16bf tmp;
            #pragma unroll
            for (int e = 0; e < 16; ++e) tmp[e] = (__bf16)src[e < 8 ? e : e + 8];
            breg[kb][jb] = tmp;
        }
}

// ---- Generic per-wave linear-recurrence scan over one 16-sequence slab --
//   Y <- Y @ Mᵀ + X_t,  X_t/out addressed as xb + t*tstride (+n*kH + jb*16)
// xb/ob/endb/initb are pre-offset by (seq0 + m0)*kH + col for this lane.
template<bool WRITE_OUT, bool WRITE_END, bool HAS_INIT>
__device__ __forceinline__ void scan_slab(const float* __restrict__ xb,
                                          size_t tstride,
                                          float* __restrict__ ob,
                                          float* __restrict__ endb,
                                          const float* __restrict__ initb,
                                          const v16bf (&breg)[4][8],
                                          __bf16* stateA,
                                          int nsteps, int lane)
{
    const int half = lane >> 4, col = lane & 15, m0 = half * 8;
    const int Lt0  = m0 + ((col & 8) ? 16 : 0);

    // seed state (WMMA A-operand bf16 layout in LDS)
    if (HAS_INIT) {
        #pragma unroll
        for (int jb = 0; jb < 8; ++jb) {
            const int kb  = jb >> 1;
            const int idx = ((jb & 1) << 3) | (col & 7);
            #pragma unroll
            for (int n = 0; n < 8; ++n)
                stateA[(kb * 32 + Lt0 + n) * 16 + idx] =
                    (__bf16)initb[(size_t)n * kH + jb * 16];
        }
    } else {
        #pragma unroll
        for (int kb = 0; kb < 4; ++kb)
            #pragma unroll
            for (int e = 0; e < 16; ++e)
                stateA[(kb * 32 + lane) * 16 + e] = (__bf16)0.0f;
    }

    for (int t = 0; t < nsteps; ++t) {
        const float* xt = xb + (size_t)t * tstride;

        if (t + 4 < nsteps)  // cover the future 8 KB slab: one line per lane
            __builtin_prefetch(xb + (size_t)(t + 4) * tstride
                               - (size_t)m0 * kH - col + lane * 64, 0, 1);

        // state -> A-operand registers (2x ds_load_b128 per kblock)
        v16bf aop[4];
        #pragma unroll
        for (int kb = 0; kb < 4; ++kb)
            aop[kb] = *(const v16bf*)&stateA[(kb * 32 + lane) * 16];

        #pragma unroll
        for (int jb = 0; jb < 8; ++jb) {
            v8f acc;
            #pragma unroll
            for (int n = 0; n < 8; ++n)
                acc[n] = xt[(size_t)n * kH + jb * 16];

            #pragma unroll
            for (int kb = 0; kb < 4; ++kb)
                acc = __builtin_amdgcn_wmma_f32_16x16x32_bf16(
                        false, aop[kb], false, breg[kb][jb],
                        (short)0, acc, false, false);

            if (WRITE_OUT) {
                float* ot = ob + (size_t)t * tstride;
                #pragma unroll
                for (int n = 0; n < 8; ++n)
                    ot[(size_t)n * kH + jb * 16] = acc[n];
            }
            if (WRITE_END && t == nsteps - 1) {
                #pragma unroll
                for (int n = 0; n < 8; ++n)
                    endb[(size_t)n * kH + jb * 16] = acc[n];
            }
            // scatter into next step's A-operand state (bf16)
            const int kb  = jb >> 1;
            const int idx = ((jb & 1) << 3) | (col & 7);
            #pragma unroll
            for (int n = 0; n < 8; ++n)
                stateA[(kb * 32 + Lt0 + n) * 16 + idx] = (__bf16)acc[n];
        }
        // single wave owns the slab: DS ops are in-order, no barrier needed
    }
}

// ---- pass0: A32 = A^32 via 5 in-LDS squarings (f32, one block) ----------
__global__ __launch_bounds__(1024)
void legs_pow(const float* __restrict__ A, float* __restrict__ A32)
{
    extern __shared__ float P[];            // 128x128 f32 = 64 KB (dynamic)
    const int tid = threadIdx.x;
    for (int i = tid; i < kH * kH; i += 1024) P[i] = A[i];
    __syncthreads();

    const int r0 = (tid >> 5) * 4;          // 4x4 register block per thread
    const int c0 = (tid & 31) * 4;
    for (int s = 0; s < 5; ++s) {
        float acc[4][4];
        #pragma unroll
        for (int i = 0; i < 4; ++i)
            #pragma unroll
            for (int j = 0; j < 4; ++j) acc[i][j] = 0.0f;
        for (int k = 0; k < kH; ++k) {
            float a0 = P[(r0 + 0) * kH + k], a1 = P[(r0 + 1) * kH + k];
            float a2 = P[(r0 + 2) * kH + k], a3 = P[(r0 + 3) * kH + k];
            v4f b = *(const v4f*)&P[k * kH + c0];
            #pragma unroll
            for (int j = 0; j < 4; ++j) {
                acc[0][j] += a0 * b[j]; acc[1][j] += a1 * b[j];
                acc[2][j] += a2 * b[j]; acc[3][j] += a3 * b[j];
            }
        }
        __syncthreads();                    // all reads done
        #pragma unroll
        for (int i = 0; i < 4; ++i) {
            v4f row = {acc[i][0], acc[i][1], acc[i][2], acc[i][3]};
            *(v4f*)&P[(r0 + i) * kH + c0] = row;
        }
        __syncthreads();                    // all writes done
    }
    for (int i = tid; i < kH * kH; i += 1024) A32[i] = P[i];
}

// ---- pass1: local chunk scans (zero init) -> end states E[c] ------------
__global__ __launch_bounds__(32, 1)
void legs_pass1(const float* __restrict__ x, const float* __restrict__ A,
                float* __restrict__ E)
{
    __shared__ __bf16 stateA[4 * 32 * 16];
    const int lane = threadIdx.x & 31;
    const int seq0 = blockIdx.x * kSlab, chunk = blockIdx.y;
    const size_t off = ((size_t)seq0 + (lane >> 4) * 8) * kH + (lane & 15);
    v16bf breg[4][8];
    load_breg(A, lane, breg);
    scan_slab<false, true, false>(x + (size_t)chunk * kChunk * kMat + off, kMat,
                                  nullptr, E + (size_t)chunk * kMat + off,
                                  nullptr, breg, stateA, kChunk, lane);
}

// ---- pass2: serial carry scan across chunks with M = A^32 ---------------
//   carry[0] = 0;  carry[c+1] = carry[c] @ (A^32)ᵀ + E[c]
__global__ __launch_bounds__(32, 1)
void legs_pass2(const float* __restrict__ E, const float* __restrict__ A32,
                float* __restrict__ carry)
{
    __shared__ __bf16 stateA[4 * 32 * 16];
    const int lane = threadIdx.x & 31;
    const int seq0 = blockIdx.x * kSlab;
    // carry[0] slab = 0
    float* c0 = carry + (size_t)seq0 * kH;
    for (int i = lane; i < kSlab * kH; i += 32) c0[i] = 0.0f;

    const size_t off = ((size_t)seq0 + (lane >> 4) * 8) * kH + (lane & 15);
    v16bf breg[4][8];
    load_breg(A32, lane, breg);
    scan_slab<true, false, false>(E + off, kMat,
                                  carry + kMat + off,  // out_t -> carry[t+1]
                                  nullptr, nullptr, breg, stateA,
                                  kNChunks - 1, lane);
}

// ---- pass3: re-run chunks seeded with carries, write final output -------
__global__ __launch_bounds__(32, 1)
void legs_pass3(const float* __restrict__ x, const float* __restrict__ A,
                const float* __restrict__ carry, float* __restrict__ out)
{
    __shared__ __bf16 stateA[4 * 32 * 16];
    const int lane = threadIdx.x & 31;
    const int seq0 = blockIdx.x * kSlab, chunk = blockIdx.y;
    const size_t off = ((size_t)seq0 + (lane >> 4) * 8) * kH + (lane & 15);
    v16bf breg[4][8];
    load_breg(A, lane, breg);
    scan_slab<true, false, true>(x + (size_t)chunk * kChunk * kMat + off, kMat,
                                 out + (size_t)chunk * kChunk * kMat + off,
                                 nullptr, carry + (size_t)chunk * kMat + off,
                                 breg, stateA, kChunk, lane);
}

// ---- fallback: fully serial single-pass scan (no workspace needed) ------
__global__ __launch_bounds__(32, 1)
void legs_serial(const float* __restrict__ x, const float* __restrict__ A,
                 float* __restrict__ out)
{
    __shared__ __bf16 stateA[4 * 32 * 16];
    const int lane = threadIdx.x & 31;
    const int seq0 = blockIdx.x * kSlab;
    const size_t off = ((size_t)seq0 + (lane >> 4) * 8) * kH + (lane & 15);
    v16bf breg[4][8];
    load_breg(A, lane, breg);
    scan_slab<true, false, false>(x + off, kMat, out + off,
                                  nullptr, nullptr, breg, stateA, kT, lane);
}

extern "C" void kernel_launch(void* const* d_in, const int* in_sizes, int n_in,
                              void* d_out, int out_size, void* d_ws, size_t ws_size,
                              hipStream_t stream)
{
    const float* x = (const float*)d_in[0];  // (T, B, N) f32
    const float* A = (const float*)d_in[1];  // (H, H) f32
    float* out = (float*)d_out;
    (void)in_sizes; (void)n_in; (void)out_size;

    if (ws_size >= kWsFloats * sizeof(float)) {
        float* ws    = (float*)d_ws;
        float* A32   = ws + kWsA32;
        float* E     = ws + kWsE;
        float* carry = ws + kWsCarry;
        legs_pass1<<<dim3(kSeq / kSlab, kNChunks), 32, 0, stream>>>(x, A, E);
        legs_pow  <<<1, 1024, kH * kH * sizeof(float), stream>>>(A, A32);
        legs_pass2<<<dim3(kSeq / kSlab), 32, 0, stream>>>(E, A32, carry);
        legs_pass3<<<dim3(kSeq / kSlab, kNChunks), 32, 0, stream>>>(x, A, carry, out);
    } else {
        legs_serial<<<dim3(kSeq / kSlab), 32, 0, stream>>>(x, A, out);
    }
}
